// CausalSelfAttention_85942295593383
// MI455X (gfx1250) — matrix operand order repaired and checked by reading, not compiled, not run on hardware
//
#include <hip/hip_runtime.h>
#include <hip/hip_bf16.h>

// ---- problem constants (match reference) ----
#define B_  2
#define N_  2048
#define D_  768
#define H_  12
#define HD_ 64
// 3*HD = 192 output cols per head in kqv

typedef __attribute__((ext_vector_type(16))) __bf16          v16bf;
typedef __attribute__((ext_vector_type(8)))  float           v8f;
typedef __attribute__((ext_vector_type(8)))  unsigned short  ushort8;

union AFrag {
    v16bf          v;       // WMMA operand
    ushort8        h[2];    // two 16-byte loads
    unsigned short u[16];
};

__device__ __forceinline__ unsigned short bf16bits(float f) {
    __hip_bfloat16 b = __float2bfloat16(f);
    union { __hip_bfloat16 b; unsigned short u; } c;
    c.b = b;
    return c.u;
}

__device__ __forceinline__ v8f wmma_bf16(v16bf a, v16bf b, v8f c) {
    // D = A(16x32 bf16) x B(32x16 bf16) + C(16x16 f32)
    return __builtin_amdgcn_wmma_f32_16x16x32_bf16(
        /*neg_a=*/false, a, /*neg_b=*/false, b,
        /*c_mod=*/(short)0, c, /*reuse_a=*/false, /*reuse_b=*/false);
}

// ---------------------------------------------------------------------------
// Kernel 0: fp32 -> bf16 conversion (grid-stride)
// ---------------------------------------------------------------------------
__global__ void __launch_bounds__(256)
convert_f32_bf16(const float* __restrict__ in, unsigned short* __restrict__ out, int n)
{
    int i = blockIdx.x * blockDim.x + threadIdx.x;
    int stride = gridDim.x * blockDim.x;
    for (; i < n; i += stride) out[i] = bf16bits(in[i]);
}

// ---------------------------------------------------------------------------
// Fragment-group loader for the two dense GEMMs: one A frag (16x32 rows of
// the activation) + six B frags (W rows, contiguous along d).
// ---------------------------------------------------------------------------
__device__ __forceinline__ void load_gemm_frags(
    AFrag& a, AFrag* bb,
    const unsigned short* __restrict__ arow,   // activation row block base
    const unsigned short* __restrict__ wbase,  // weight base [*,D_]
    int k0, int lm, int lh, int col0)
{
    const unsigned short* ap = arow + (size_t)lm * D_ + k0 + 8 * lh;
    a.h[0] = *reinterpret_cast<const ushort8*>(ap);
    a.h[1] = *reinterpret_cast<const ushort8*>(ap + 16);
    #pragma unroll
    for (int t = 0; t < 6; ++t) {
        const int o = col0 + t * 16 + lm;
        const unsigned short* bp = wbase + (size_t)o * D_ + k0 + 8 * lh;
        bb[t].h[0] = *reinterpret_cast<const ushort8*>(bp);
        bb[t].h[1] = *reinterpret_cast<const ushort8*>(bp + 16);
    }
}

// ---------------------------------------------------------------------------
// Kernel 1: KQV projection (double-buffered k-loop).  For each (b,h):
//   O[n, o] = sum_d X[n,d] * Wkqv[h,o,d] + b[h,o]      (o in [0,192))
//   o in [0,64)    -> K  stored [B,H,N,64]  bf16
//   o in [64,128)  -> Q  stored [B,H,N,64]  bf16, pre-scaled by 1/sqrt(64)
//   o in [128,192) -> V  stored TRANSPOSED [B,H,64,N] bf16
// ---------------------------------------------------------------------------
__global__ void __launch_bounds__(256)
kqv_kernel(const unsigned short* __restrict__ xb,     // [B,N,D] bf16
           const unsigned short* __restrict__ wb,     // [H,192,D] bf16
           const float*          __restrict__ bias,   // [H,192]
           unsigned short* __restrict__ Km,           // [B,H,N,HD]
           unsigned short* __restrict__ Qm,           // [B,H,N,HD]
           unsigned short* __restrict__ Vt)           // [B,H,HD,N]
{
    const int lane = threadIdx.x & 31;
    const int wave = threadIdx.x >> 5;
    const int lm = lane & 15, lh = lane >> 4;
    const int b = blockIdx.z, h = blockIdx.y;
    const int row0 = blockIdx.x * 64 + (wave >> 1) * 16;
    const int col0 = (wave & 1) * 96;

    const unsigned short* xrow  = xb + ((size_t)b * N_ + row0) * D_;
    const unsigned short* wbase = wb + (size_t)h * 192 * D_;

    v8f zero = {};
    v8f acc[6];
    for (int t = 0; t < 6; ++t) acc[t] = zero;

    AFrag a[2], bb[2][6];
    load_gemm_frags(a[0], bb[0], xrow, wbase, 0, lm, lh, col0);

    int buf = 0;
    for (int k0 = 0; k0 < D_; k0 += 32, buf ^= 1) {
        if (k0 + 32 < D_)  // prefetch next k-step into the other buffer
            load_gemm_frags(a[buf ^ 1], bb[buf ^ 1], xrow, wbase, k0 + 32, lm, lh, col0);
        #pragma unroll
        for (int t = 0; t < 6; ++t)
            acc[t] = wmma_bf16(a[buf].v, bb[buf][t].v, acc[t]);
    }

    const float qscale = 0.125f;  // 1/sqrt(HD), folded into Q
    const size_t bh = (size_t)b * H_ + h;
    #pragma unroll
    for (int t = 0; t < 6; ++t) {
        const int o = col0 + t * 16 + lm;
        const float bv = bias[h * 192 + o];
        #pragma unroll
        for (int r = 0; r < 8; ++r) {
            const int n = row0 + r + 8 * lh;       // C layout: row = r + 8*(lane/16)
            const float val = acc[t][r] + bv;
            if (o < 64) {
                Km[(bh * N_ + n) * HD_ + o] = bf16bits(val);
            } else if (o < 128) {
                Qm[(bh * N_ + n) * HD_ + (o - 64)] = bf16bits(val * qscale);
            } else {
                Vt[(bh * HD_ + (o - 128)) * N_ + n] = bf16bits(val);
            }
        }
    }
}

// ---------------------------------------------------------------------------
// Kernel 2: flash attention.  Workgroup = 8 waves, each owns 16 query rows.
// Per 32-key block: load K frags + V frags up front (V latency hidden by the
// softmax VALU work), S via 4 WMMAs, online softmax, P through per-wave LDS
// tile, O += P @ V via 4 WMMAs.
// ---------------------------------------------------------------------------
__global__ void __launch_bounds__(256)
attn_kernel(const unsigned short* __restrict__ Qm,  // [B,H,N,HD] (pre-scaled)
            const unsigned short* __restrict__ Km,  // [B,H,N,HD]
            const unsigned short* __restrict__ Vt,  // [B,H,HD,N]
            unsigned short*       __restrict__ SA)  // [B,N,D] bf16 (heads concat)
{
    __shared__ unsigned short pbuf[8][16][40];  // per-wave 16x32 P tile, padded

    const int lane = threadIdx.x & 31;
    const int wave = threadIdx.x >> 5;
    const int lm = lane & 15, lh = lane >> 4;
    const int b = blockIdx.z, h = blockIdx.y;
    const int q0 = blockIdx.x * 128 + wave * 16;

    const size_t bh = (size_t)b * H_ + h;
    const unsigned short* qp = Qm + (bh * N_ + q0) * HD_;
    const unsigned short* kp = Km + bh * N_ * HD_;
    const unsigned short* vp = Vt + bh * HD_ * N_;

    // Q tile (16x64) -> two A fragments, loaded once
    AFrag qf[2];
    #pragma unroll
    for (int kk = 0; kk < 2; ++kk) {
        const unsigned short* ap = qp + (size_t)lm * HD_ + kk * 32 + 8 * lh;
        qf[kk].h[0] = *reinterpret_cast<const ushort8*>(ap);
        qf[kk].h[1] = *reinterpret_cast<const ushort8*>(ap + 16);
    }

    v8f zero = {};
    v8f o[4];
    for (int t = 0; t < 4; ++t) o[t] = zero;
    float mrow[8], lrow[8];
    #pragma unroll
    for (int r = 0; r < 8; ++r) { mrow[r] = -1e30f; lrow[r] = 0.f; }

    const int qmax = q0 + 15;
    for (int kb = 0; kb <= qmax; kb += 32) {
        // ---- issue ALL loads for this block first: K frags + V frags ----
        AFrag kf[2][2];   // [k-step][key tile]; B frag: col = key, run over d
        #pragma unroll
        for (int kk = 0; kk < 2; ++kk) {
            const unsigned short* bp0 = kp + (size_t)(kb + lm)      * HD_ + kk * 32 + 8 * lh;
            const unsigned short* bp1 = kp + (size_t)(kb + 16 + lm) * HD_ + kk * 32 + 8 * lh;
            kf[kk][0].h[0] = *reinterpret_cast<const ushort8*>(bp0);
            kf[kk][0].h[1] = *reinterpret_cast<const ushort8*>(bp0 + 16);
            kf[kk][1].h[0] = *reinterpret_cast<const ushort8*>(bp1);
            kf[kk][1].h[1] = *reinterpret_cast<const ushort8*>(bp1 + 16);
        }
        AFrag vf[4];      // B frag: col = dim (t*16+lm), run over keys (Vt rows)
        #pragma unroll
        for (int t = 0; t < 4; ++t) {
            const unsigned short* vrow = vp + (size_t)(t * 16 + lm) * N_ + kb + 8 * lh;
            vf[t].h[0] = *reinterpret_cast<const ushort8*>(vrow);
            vf[t].h[1] = *reinterpret_cast<const ushort8*>(vrow + 16);
        }

        // ---- S = Q @ K^T : two 16x16 tiles (keys kb..+15 and kb+16..+31) ----
        v8f s0 = zero, s1 = zero;
        #pragma unroll
        for (int kk = 0; kk < 2; ++kk) {
            s0 = wmma_bf16(qf[kk].v, kf[kk][0].v, s0);
            s1 = wmma_bf16(qf[kk].v, kf[kk][1].v, s1);
        }

        // ---- causal mask + online softmax (row stats via 16-lane shfl) ----
        // V loads complete underneath this VALU-heavy section.
        float p0[8], p1[8];
        #pragma unroll
        for (int r = 0; r < 8; ++r) {
            const int q    = q0 + r + 8 * lh;   // this lane's row for VGPR r
            const int key0 = kb + lm;
            const int key1 = kb + 16 + lm;
            float v0 = (key0 <= q) ? s0[r] : -1e30f;
            float v1 = (key1 <= q) ? s1[r] : -1e30f;
            float rmax = fmaxf(v0, v1);
            #pragma unroll
            for (int off = 1; off < 16; off <<= 1)
                rmax = fmaxf(rmax, __shfl_xor(rmax, off, 32));
            const float mnew  = fmaxf(mrow[r], rmax);
            const float alpha = __expf(mrow[r] - mnew);
            const float e0 = (key0 <= q) ? __expf(v0 - mnew) : 0.f;
            const float e1 = (key1 <= q) ? __expf(v1 - mnew) : 0.f;
            float rsum = e0 + e1;
            #pragma unroll
            for (int off = 1; off < 16; off <<= 1)
                rsum += __shfl_xor(rsum, off, 32);
            lrow[r] = lrow[r] * alpha + rsum;
            mrow[r] = mnew;
            #pragma unroll
            for (int t = 0; t < 4; ++t) o[t][r] *= alpha;
            p0[r] = e0; p1[r] = e1;
        }

        // ---- P (C layout, f32) -> bf16 A fragment via per-wave LDS tile ----
        #pragma unroll
        for (int r = 0; r < 8; ++r) {
            pbuf[wave][r + 8 * lh][lm]      = bf16bits(p0[r]);
            pbuf[wave][r + 8 * lh][16 + lm] = bf16bits(p1[r]);
        }
        AFrag pf;  // row = lm, key runs [8lh,+7] and [16+8lh,+7]
        pf.h[0] = *reinterpret_cast<const ushort8*>(&pbuf[wave][lm][8 * lh]);
        pf.h[1] = *reinterpret_cast<const ushort8*>(&pbuf[wave][lm][16 + 8 * lh]);

        // ---- O += P @ V ----
        #pragma unroll
        for (int t = 0; t < 4; ++t)
            o[t] = wmma_bf16(pf.v, vf[t].v, o[t]);
    }

    // ---- finalize: O /= l, write concat-head bf16 SA[b,n,h*64+dim] ----
    #pragma unroll
    for (int r = 0; r < 8; ++r) {
        const float inv = 1.f / lrow[r];
        const int q = q0 + r + 8 * lh;
        #pragma unroll
        for (int t = 0; t < 4; ++t) {
            const int dcol = h * HD_ + t * 16 + lm;
            SA[((size_t)b * N_ + q) * D_ + dcol] = bf16bits(o[t][r] * inv);
        }
    }
}

// ---------------------------------------------------------------------------
// Kernel 3: output projection (double-buffered k-loop)
//   out[n,o] = sum_d SA[n,d]*Wproj[o,d] + b[o]
// ---------------------------------------------------------------------------
__global__ void __launch_bounds__(256)
proj_kernel(const unsigned short* __restrict__ SAb,  // [B*N, D] bf16
            const unsigned short* __restrict__ Wb,   // [D, D]  bf16
            const float*          __restrict__ bias, // [D]
            float*                __restrict__ out)  // [B*N, D] f32
{
    const int lane = threadIdx.x & 31;
    const int wave = threadIdx.x >> 5;
    const int lm = lane & 15, lh = lane >> 4;
    const int row0 = blockIdx.x * 64 + (wave >> 1) * 16;
    const int col0 = blockIdx.y * 192 + (wave & 1) * 96;

    v8f zero = {};
    v8f acc[6];
    for (int t = 0; t < 6; ++t) acc[t] = zero;

    const unsigned short* arow = SAb + (size_t)row0 * D_;

    AFrag a[2], bb[2][6];
    load_gemm_frags(a[0], bb[0], arow, Wb, 0, lm, lh, col0);

    int buf = 0;
    for (int k0 = 0; k0 < D_; k0 += 32, buf ^= 1) {
        if (k0 + 32 < D_)
            load_gemm_frags(a[buf ^ 1], bb[buf ^ 1], arow, Wb, k0 + 32, lm, lh, col0);
        #pragma unroll
        for (int t = 0; t < 6; ++t)
            acc[t] = wmma_bf16(a[buf].v, bb[buf][t].v, acc[t]);
    }

    #pragma unroll
    for (int t = 0; t < 6; ++t) {
        const int o = col0 + t * 16 + lm;
        const float bv = bias[o];
        #pragma unroll
        for (int r = 0; r < 8; ++r) {
            const int n = row0 + r + 8 * lh;
            out[(size_t)n * D_ + o] = acc[t][r] + bv;
        }
    }
}

// ---------------------------------------------------------------------------
// Launcher
// ---------------------------------------------------------------------------
extern "C" void kernel_launch(void* const* d_in, const int* in_sizes, int n_in,
                              void* d_out, int out_size, void* d_ws, size_t ws_size,
                              hipStream_t stream) {
    const float* x     = (const float*)d_in[0];   // [B,N,D]
    const float* Wkqv  = (const float*)d_in[1];   // [H,192,D]
    const float* bkqv  = (const float*)d_in[2];   // [H,192]
    const float* Wproj = (const float*)d_in[3];   // [D,D]
    const float* bproj = (const float*)d_in[4];   // [D]
    float* out = (float*)d_out;

    // workspace layout (bf16 elements), all chunks >=16B aligned
    const size_t nx  = (size_t)B_ * N_ * D_;        // 3,145,728
    const size_t nwk = (size_t)H_ * 3 * HD_ * D_;   // 1,769,472
    const size_t nwp = (size_t)D_ * D_;             //   589,824
    const size_t nhd = (size_t)B_ * H_ * N_ * HD_;  // 3,145,728

    unsigned short* ws     = (unsigned short*)d_ws;
    unsigned short* xb     = ws;
    unsigned short* wkqvb  = xb + nx;
    unsigned short* wprojb = wkqvb + nwk;
    unsigned short* Km     = wprojb + nwp;
    unsigned short* Qm     = Km + nhd;
    unsigned short* Vt     = Qm + nhd;
    unsigned short* SAb    = Vt + nhd;
    // total: ~36.2 MB of d_ws

    convert_f32_bf16<<<dim3((unsigned)((nx  + 255) / 256)), 256, 0, stream>>>(x,     xb,     (int)nx);
    convert_f32_bf16<<<dim3((unsigned)((nwk + 255) / 256)), 256, 0, stream>>>(Wkqv,  wkqvb,  (int)nwk);
    convert_f32_bf16<<<dim3((unsigned)((nwp + 255) / 256)), 256, 0, stream>>>(Wproj, wprojb, (int)nwp);

    kqv_kernel <<<dim3(N_ / 64, H_, B_),        256, 0, stream>>>(xb, wkqvb, bkqv, Km, Qm, Vt);
    attn_kernel<<<dim3(N_ / 128, H_, B_),       256, 0, stream>>>(Qm, Km, Vt, SAb);
    proj_kernel<<<dim3(B_ * N_ / 64, D_ / 192), 256, 0, stream>>>(SAb, wprojb, bproj, out);
}
